// SmallMLP_INR_47244640256091
// MI455X (gfx1250) — matrix operand order repaired and checked by reading, not compiled, not run on hardware
//
#include <hip/hip_runtime.h>

typedef __attribute__((ext_vector_type(16))) _Float16 v16h;
typedef __attribute__((ext_vector_type(8)))  _Float16 v8h;
typedef __attribute__((ext_vector_type(8)))  float    v8f;

#define XSTRIDE 264   // 256 + 8 halfword pad -> conflict-free ds_load_b128 on A operand

// ---------------------------------------------------------------------------
// Repack W2..W5 (fp32, 256x256 row-major, W[k][n]) into a
// V_WMMA_F32_16X16X32_F16 B-operand lane layout (contiguous k = 16*(L>>4) + j), f16, into d_ws.
// Tile (l, kt, nt): 32(K) x 16(N), 1024 bytes, laid out as two 512B halves:
//   halfword index = l*65536 + (kt*16+nt)*512 + (j>>3)*256 + L*8 + (j&7)
// where lane L in [0,32), j in [0,16), K = kt*32 + (L>>4)*16 + j, N = nt*16 + (L&15).
// Chunk c of a layer = K-tiles {2c, 2c+1} x all 16 N-tiles = contiguous 32 KB.
// ---------------------------------------------------------------------------
__global__ __launch_bounds__(256) void repack_w_kernel(
    const float* __restrict__ W2, const float* __restrict__ W3,
    const float* __restrict__ W4, const float* __restrict__ W5,
    unsigned short* __restrict__ wsOut)
{
  int f = blockIdx.x * 256 + threadIdx.x;      // 0 .. 262143
  if (f >= 4 * 65536) return;
  int l      = f >> 16;
  int rem    = f & 65535;
  int tile   = rem >> 9;                        // 0..127
  int within = rem & 511;                       // 0..511
  int kt = tile >> 4, nt = tile & 15;
  int L  = within >> 4, j = within & 15;
  const float* W = (l == 0) ? W2 : (l == 1) ? W3 : (l == 2) ? W4 : W5;
  int K  = kt * 32 + (L >> 4) * 16 + j;
  int Nc = nt * 16 + (L & 15);
  _Float16 hv = (_Float16)W[K * 256 + Nc];
  int dst = l * 65536 + tile * 512 + (j >> 3) * 256 + L * 8 + (j & 7);
  wsOut[dst] = __builtin_bit_cast(unsigned short, hv);
}

// ---------------------------------------------------------------------------
// Issue one 32KB weight chunk as async global->LDS copies (no VGPR roundtrip,
// tracked by ASYNCcnt). 256 threads x 8 x b128 = 32 KB.
// ---------------------------------------------------------------------------
__device__ __forceinline__ void issue_async_chunk(const char* src,  // uniform
                                                  unsigned lds_base_b,
                                                  int tid)
{
  unsigned v0 = (unsigned)(tid * 16);
#pragma unroll
  for (int t = 0; t < 8; ++t) {
    unsigned ldsb = lds_base_b + (unsigned)(t * 4096) + (unsigned)(tid * 16);
    unsigned voff = v0 + (unsigned)(t * 4096);
    asm volatile("global_load_async_to_lds_b128 %0, %1, %2"
                 :
                 : "v"(ldsb), "v"(voff), "s"(src)
                 : "memory");
  }
}

// ---------------------------------------------------------------------------
// Fused MLP: 128 rows per workgroup, 8 waves, each wave owns a 64x64 output
// block (4x4 WMMA tiles) per hidden layer. Activations live in LDS (f16),
// updated in place layer by layer. Weights stream through a ping-pong 32 KB
// LDS pair via async-to-LDS copies overlapped with WMMA compute.
// ---------------------------------------------------------------------------
__global__ __launch_bounds__(256) void mlp_fused_kernel(
    const float* __restrict__ coords,
    const float* __restrict__ W1, const float* __restrict__ b1,
    const float* __restrict__ b2, const float* __restrict__ b3,
    const float* __restrict__ b4, const float* __restrict__ b5,
    const float* __restrict__ W6, const float* __restrict__ b6,
    const unsigned short* __restrict__ wsW,
    float* __restrict__ out, int N)
{
  __shared__ __align__(16) _Float16 sX[128 * XSTRIDE];  // 67,584 B activations
  __shared__ __align__(16) _Float16 sW[64 * 512];       // 2 x 32 KB weight chunks

  const int tid  = threadIdx.x;
  const int lane = tid & 31;
  const int wv   = tid >> 5;
  const int rowBase = blockIdx.x * 128;
  const char* wsBytes = (const char*)wsW;
  const unsigned sWbase32 = (unsigned)(uintptr_t)(void*)sW;  // LDS byte offset

  // Prologue: start streaming chunk 0 (layer 2, kt 0..1) immediately so it
  // overlaps the VALU input layer below.
  issue_async_chunk(wsBytes, sWbase32, tid);

  // ---- Layer 1: x = relu(coords @ W1 + b1), write f16 activations to LDS ----
  {
    int r = tid >> 1, hf = tid & 1;
    int grow = rowBase + r;
    float c0 = 0.f, c1 = 0.f;
    if (grow < N) { c0 = coords[2 * grow]; c1 = coords[2 * grow + 1]; }
    int col0 = hf * 128;
#pragma unroll 4
    for (int j = 0; j < 128; ++j) {
      int col = col0 + j;
      float v = fmaf(c1, W1[256 + col], fmaf(c0, W1[col], b1[col]));
      sX[r * XSTRIDE + col] = (_Float16)fmaxf(v, 0.f);
    }
  }
  __syncthreads();

  const int mtg  = (wv >> 2) * 64;     // wave's row-block base (0 or 64)
  const int ntg  = (wv & 3) * 4;       // wave's first N-tile (0,4,8,12)
  const int lrow = lane & 15;
  const int lhi  = lane >> 4;
  const float* biases[4] = { b2, b3, b4, b5 };
  const v8f vzero = {0.f, 0.f, 0.f, 0.f, 0.f, 0.f, 0.f, 0.f};

  for (int l = 0; l < 4; ++l) {
    float bv[4];
#pragma unroll
    for (int i = 0; i < 4; ++i) bv[i] = biases[l][(ntg + i) * 16 + lrow];

    v8f acc[4][4];
#pragma unroll
    for (int mt = 0; mt < 4; ++mt)
#pragma unroll
      for (int i = 0; i < 4; ++i) acc[mt][i] = vzero;

    for (int c = 0; c < 4; ++c) {
      const int g = l * 4 + c;               // global chunk id, 0..15

      // own wave's async copies for chunk g have completed
      asm volatile("s_wait_asynccnt 0x0" ::: "memory");
      // all waves: chunk g staged, and everyone is done reading buffer (g+1)&1
      __syncthreads();

      // stream chunk g+1 into the other half while we compute chunk g
      if (g + 1 < 16) {
        issue_async_chunk(wsBytes + (size_t)(g + 1) * 32768,
                          sWbase32 + (unsigned)(((g + 1) & 1) * 32768), tid);
      }
      // warm chunk g+2 toward L2 (global_prefetch_b8)
      if (g + 2 < 16) {
        __builtin_prefetch(wsBytes + (size_t)(g + 2) * 32768 + tid * 128, 0, 0);
      }

      const int bufE = (g & 1) * 16384;      // element offset of live buffer
#pragma unroll
      for (int ktl = 0; ktl < 2; ++ktl) {
        const int kg = c * 2 + ktl;          // global K-tile (K base = kg*32)
        // A operands: wave's 4 row tiles, K-slice kg*32..+31, from sX
        v16h A[4];
#pragma unroll
        for (int mt = 0; mt < 4; ++mt) {
          const _Float16* p = &sX[(mtg + mt * 16 + lrow) * XSTRIDE + kg * 32 + lhi * 16];
          v8h lo = *(const v8h*)p;          // K = base .. base+7
          v8h hi = *(const v8h*)(p + 8);    // K = base+8 .. base+15
          A[mt] = __builtin_shufflevector(lo, hi,
                    0, 1, 2, 3, 4, 5, 6, 7, 8, 9, 10, 11, 12, 13, 14, 15);
        }
        // B operands: 4 N-tiles, loaded up front so all 16 ds_loads are in
        // flight before the first dependent WMMA
        v16h B[4];
#pragma unroll
        for (int i = 0; i < 4; ++i) {
          const _Float16* p = &sW[bufE + (ktl * 16 + ntg + i) * 512 + lane * 8];
          v8h lo = *(const v8h*)p;
          v8h hi = *(const v8h*)(p + 256);
          B[i] = __builtin_shufflevector(lo, hi,
                   0, 1, 2, 3, 4, 5, 6, 7, 8, 9, 10, 11, 12, 13, 14, 15);
        }
#pragma unroll
        for (int i = 0; i < 4; ++i) {
#pragma unroll
          for (int mt = 0; mt < 4; ++mt) {
            acc[mt][i] = __builtin_amdgcn_wmma_f32_16x16x32_f16(
                false, A[mt], false, B[i], (short)0, acc[mt][i], false, false);
          }
        }
      }
    }

    // all waves done reading sX for this layer before in-place update
    __syncthreads();

    // ---- bias + ReLU, convert to f16, store activations back in place ----
#pragma unroll
    for (int mt = 0; mt < 4; ++mt) {
#pragma unroll
      for (int i = 0; i < 4; ++i) {
        const int col = (ntg + i) * 16 + lrow;
#pragma unroll
        for (int v = 0; v < 8; ++v) {
          float r = fmaxf(acc[mt][i][v] + bv[i], 0.f);
          int row = mtg + mt * 16 + v + lhi * 8;   // C/D layout: M = v + (L>=16)*8
          sX[row * XSTRIDE + col] = (_Float16)r;
        }
      }
    }
    __syncthreads();
  }

  // ---- Layer 6: out = x @ W6 + b6 (dot of 256), wave32 pair reduction ----
  {
    int r = tid >> 1, hf = tid & 1;
    int grow = rowBase + r;
    float acc = 0.f;
    int col0 = hf * 128;
#pragma unroll 4
    for (int j = 0; j < 128; ++j) {
      int col = col0 + j;
      acc = fmaf((float)sX[r * XSTRIDE + col], W6[col], acc);
    }
    float tot = acc + __shfl_xor(acc, 1, 32);
    if (hf == 0 && grow < N) out[grow] = tot + b6[0];
  }
}

// ---------------------------------------------------------------------------
extern "C" void kernel_launch(void* const* d_in, const int* in_sizes, int n_in,
                              void* d_out, int out_size, void* d_ws, size_t ws_size,
                              hipStream_t stream) {
  (void)n_in; (void)out_size; (void)ws_size;
  const float* coords = (const float*)d_in[0];
  const float* W1 = (const float*)d_in[1];
  const float* b1 = (const float*)d_in[2];
  const float* W2 = (const float*)d_in[3];
  const float* b2 = (const float*)d_in[4];
  const float* W3 = (const float*)d_in[5];
  const float* b3 = (const float*)d_in[6];
  const float* W4 = (const float*)d_in[7];
  const float* b4 = (const float*)d_in[8];
  const float* W5 = (const float*)d_in[9];
  const float* b5 = (const float*)d_in[10];
  const float* W6 = (const float*)d_in[11];
  const float* b6 = (const float*)d_in[12];
  float* out = (float*)d_out;
  unsigned short* ws16 = (unsigned short*)d_ws;   // 512 KB of repacked f16 weights

  const int N = in_sizes[0] / 2;

  // 1) repack hidden-layer weights into WMMA B-operand layout (f16)
  repack_w_kernel<<<(4 * 65536 + 255) / 256, 256, 0, stream>>>(W2, W3, W4, W5, ws16);

  // 2) fused 6-layer MLP, 128 rows per workgroup
  const int blocks = (N + 127) / 128;
  mlp_fused_kernel<<<blocks, 256, 0, stream>>>(
      coords, W1, b1, b2, b3, b4, b5, W6, b6, ws16, out, N);
}
